// EGNN_26096221290525
// MI455X (gfx1250) — matrix operand order repaired and checked by reading, not compiled, not run on hardware
//
#include <hip/hip_runtime.h>

// ---------------------------------------------------------------------------
// EGNN forward for MI455X (gfx1250, wave32, WMMA + async global->LDS).
// N=50000 nodes, E=800000 edges, HID=128, 4 layers.
// ---------------------------------------------------------------------------

#define N_NODES 50000
#define N_EDGES 800000
#define HID 128

typedef _Float16 v16h __attribute__((ext_vector_type(16)));
typedef float    v8f  __attribute__((ext_vector_type(8)));

// gfx1250 async global->LDS DMA (ASYNCcnt tracked), with portable fallback.
#ifndef USE_ASYNC_LDS
#if defined(__has_builtin)
#if __has_builtin(__builtin_amdgcn_global_load_async_to_lds_b128) && \
    __has_builtin(__builtin_amdgcn_s_wait_asynccnt)
#define USE_ASYNC_LDS 1
#endif
#endif
#endif
#ifndef USE_ASYNC_LDS
#define USE_ASYNC_LDS 0
#endif

#if USE_ASYNC_LDS
typedef int i32x4 __attribute__((ext_vector_type(4)));
typedef __attribute__((address_space(1))) i32x4* gv4p;  // global int4*
typedef __attribute__((address_space(3))) i32x4* lv4p;  // LDS int4*
#endif

// 16-byte global -> LDS copy.
__device__ __forceinline__ void cp16(void* dst_lds, const void* src_g) {
#if USE_ASYNC_LDS
  __builtin_amdgcn_global_load_async_to_lds_b128((gv4p)src_g, (lv4p)dst_lds,
                                                 0, 0);
#else
  *(uint4*)dst_lds = *(const uint4*)src_g;
#endif
}
__device__ __forceinline__ void cp_fence() {
#if USE_ASYNC_LDS
  __builtin_amdgcn_s_wait_asynccnt(0);
#endif
}

__device__ __forceinline__ v8f wmma_f16(v16h a, v16h b, v8f c) {
  // D = A(16x32 f16) * B(32x16 f16) + C(16x16 f32)
  return __builtin_amdgcn_wmma_f32_16x16x32_f16(false, a, false, b,
                                                (short)0, c, false, false);
}

// A fragment: 16x32 tile, row-major, leading dim `ld` (LDS resident).
__device__ __forceinline__ v16h frag_a(const _Float16* tile, int ld, int lane) {
  int m = lane & 15, g = lane >> 4;
  const _Float16* row = tile + m * ld;
  v16h a;
#pragma unroll
  for (int v = 0; v < 8; ++v) {
    int k = (v < 4) ? (g * 8 + 2 * v) : (16 + g * 8 + 2 * (v - 4));
    a[2 * v]     = row[k];
    a[2 * v + 1] = row[k + 1];
  }
  return a;
}

// B fragment: 32x16 tile of row-major [K][ld] weights (global, L2-hot).
__device__ __forceinline__ v16h frag_b(const _Float16* tile, int ld, int lane) {
  int n = lane & 15, g = lane >> 4;
  v16h b;
#pragma unroll
  for (int v = 0; v < 8; ++v) {
    int k = (v < 4) ? (g * 8 + 2 * v) : (16 + g * 8 + 2 * (v - 4));
    b[2 * v]     = tile[(size_t)k * ld + n];
    b[2 * v + 1] = tile[(size_t)(k + 1) * ld + n];
  }
  return b;
}

__device__ __forceinline__ float fast_sigmoid(float x) {
  return __builtin_amdgcn_rcpf(1.0f + __expf(-x));   // v_rcp_f32, no div chain
}
__device__ __forceinline__ float silu_f(float x) { return x * fast_sigmoid(x); }

// --------------------------- fp32 -> fp16 weight convert (K zero-padded) ---
__global__ __launch_bounds__(256) void k_cvt_f16(const float* __restrict__ src,
                                                 _Float16* __restrict__ dst,
                                                 int K, int Kpad, int Ncol) {
  long long gid = (long long)blockIdx.x * 256 + threadIdx.x;
  long long tot = (long long)Kpad * Ncol;
  if (gid >= tot) return;
  int k = (int)(gid / Ncol);
  int n = (int)(gid - (long long)k * Ncol);
  dst[gid] = (k < K) ? (_Float16)src[(long long)k * Ncol + n] : (_Float16)0.0f;
}

// --------------------------- edge pass 1: fused edge MLP -------------------
// 32 edges per workgroup, 8 waves. e_in K = 258 padded to 288 (ld 296).
#define TE 32
#define LDA 296
#define LDH 136
__global__ __launch_bounds__(256) void k_edge_pass1(
    const _Float16* __restrict__ hbf, const float* __restrict__ coords,
    const int* __restrict__ eidx, const float* __restrict__ eattr,
    const _Float16* __restrict__ We1, const float* __restrict__ be1,
    const _Float16* __restrict__ We2, const float* __restrict__ be2,
    const _Float16* __restrict__ Wc1, const float* __restrict__ bc1,
    const float* __restrict__ attW, const float* __restrict__ attB,
    const float* __restrict__ c2W,
    _Float16* __restrict__ m_out, float* __restrict__ expa,
    float* __restrict__ cwv, float* __restrict__ dsum, int nE) {
  __shared__ _Float16 eA[TE][LDA];   // e_in, K padded
  __shared__ _Float16 sm1[TE][LDH];  // silu(e1)
  __shared__ _Float16 sm2[TE][LDH];  // silu(e2) = m
  __shared__ _Float16 sm3[TE][LDH];  // silu(c1)
  __shared__ int sdst[TE];

  const int t  = threadIdx.x;
  const int eb = blockIdx.x * TE;

  { // stage e_in: 8 threads per edge, 32B of h[dst] + 32B of h[src] each
    int el = t >> 3, q = t & 7;
    int e = eb + el; if (e >= nE) e = nE - 1;
    int s = eidx[e], d = eidx[nE + e];
    const _Float16* hd = hbf + (size_t)d * HID + q * 16;
    const _Float16* hs = hbf + (size_t)s * HID + q * 16;
    cp16(&eA[el][q * 16], hd);
    cp16(&eA[el][q * 16 + 8], hd + 8);
    cp16(&eA[el][HID + q * 16], hs);
    cp16(&eA[el][HID + q * 16 + 8], hs + 8);
    if (q == 0) {
      float rx = coords[d * 3 + 0] - coords[s * 3 + 0];
      float ry = coords[d * 3 + 1] - coords[s * 3 + 1];
      float rz = coords[d * 3 + 2] - coords[s * 3 + 2];
      eA[el][256] = (_Float16)(rx * rx + ry * ry + rz * rz);
      eA[el][257] = (_Float16)eattr[e];
#pragma unroll
      for (int i = 258; i < 288; ++i) eA[el][i] = (_Float16)0.0f;
      sdst[el] = d;
    }
  }
  cp_fence();
  __syncthreads();

  const int wave = t >> 5, lane = t & 31;
  const int n0 = wave * 16, nn = n0 + (lane & 15), g = lane >> 4;

  // GEMM1: e_in[32x288] @ We1[288x128] -> silu -> sm1
  for (int rt = 0; rt < TE / 16; ++rt) {
    v8f acc = {};
#pragma unroll
    for (int kt = 0; kt < 9; ++kt)
      acc = wmma_f16(frag_a(&eA[rt * 16][kt * 32], LDA, lane),
                     frag_b(We1 + (size_t)(kt * 32) * HID + n0, HID, lane), acc);
    float bias = be1[nn];
#pragma unroll
    for (int r = 0; r < 8; ++r)
      sm1[rt * 16 + r + 8 * g][nn] = (_Float16)silu_f(acc[r] + bias);
  }
  __syncthreads();

  // GEMM2: sm1 @ We2 -> silu -> sm2
  for (int rt = 0; rt < TE / 16; ++rt) {
    v8f acc = {};
#pragma unroll
    for (int kt = 0; kt < 4; ++kt)
      acc = wmma_f16(frag_a(&sm1[rt * 16][kt * 32], LDH, lane),
                     frag_b(We2 + (size_t)(kt * 32) * HID + n0, HID, lane), acc);
    float bias = be2[nn];
#pragma unroll
    for (int r = 0; r < 8; ++r)
      sm2[rt * 16 + r + 8 * g][nn] = (_Float16)silu_f(acc[r] + bias);
  }
  __syncthreads();

  { // spill m to HBM as wide 16B stores (read back in pass 2)
    int el = t >> 3, q = t & 7;
    int e = eb + el;
    if (e < nE) {
      const uint4* s = (const uint4*)&sm2[el][q * 16];
      uint4* dp = (uint4*)(m_out + (size_t)e * HID + q * 16);
      dp[0] = s[0];
      dp[1] = s[1];
    }
  }

  // GEMM3: sm2 @ Wc1 -> silu -> sm3
  for (int rt = 0; rt < TE / 16; ++rt) {
    v8f acc = {};
#pragma unroll
    for (int kt = 0; kt < 4; ++kt)
      acc = wmma_f16(frag_a(&sm2[rt * 16][kt * 32], LDH, lane),
                     frag_b(Wc1 + (size_t)(kt * 32) * HID + n0, HID, lane), acc);
    float bias = bc1[nn];
#pragma unroll
    for (int r = 0; r < 8; ++r)
      sm3[rt * 16 + r + 8 * g][nn] = (_Float16)silu_f(acc[r] + bias);
  }
  __syncthreads();

  // Per-edge att & c2 dot products: 8 lanes/edge + shfl_xor tree reduce.
  // att logit -> sigmoid in (0,1) -> exp needs no segment-max stabilization.
  {
    int el = wave * 4 + ((lane >> 3) & 3);  // 4 edges per wave
    int fp = (lane & 7) * 16;
    float da = 0.0f, dc = 0.0f;
#pragma unroll
    for (int i = 0; i < 16; ++i) {
      da += (float)sm2[el][fp + i] * attW[fp + i];
      dc += (float)sm3[el][fp + i] * c2W[fp + i];
    }
#pragma unroll
    for (int s = 4; s > 0; s >>= 1) {
      da += __shfl_xor(da, s, 32);
      dc += __shfl_xor(dc, s, 32);
    }
    if ((lane & 7) == 0) {
      int e = eb + el;
      if (e < nE) {
        float a  = fast_sigmoid(da + attB[0]);
        float ex = __expf(a);
        expa[e] = ex;
        cwv[e]  = dc;
        atomicAdd(&dsum[sdst[el]], ex);
      }
    }
  }
}

// --------------------------- edge pass 2: alpha-weighted scatter -----------
__global__ __launch_bounds__(256) void k_edge_pass2(
    const _Float16* __restrict__ m_buf, const float* __restrict__ expa,
    const float* __restrict__ cwv, const float* __restrict__ dsum,
    const int* __restrict__ eidx, const float* __restrict__ coords,
    float* __restrict__ m_i, float* __restrict__ cacc, int nE) {
  long long gid = (long long)blockIdx.x * 256 + threadIdx.x;
  int e = (int)(gid >> 5), l = (int)(gid & 31);
  if (e >= nE) return;
  int s = eidx[e], d = eidx[nE + e];
  float alpha = expa[e] * __builtin_amdgcn_rcpf(dsum[d] + 1e-16f);
  const _Float16* mrow = m_buf + (size_t)e * HID + l * 4;
  float* acc = m_i + (size_t)d * HID + l * 4;
#pragma unroll
  for (int i = 0; i < 4; ++i) atomicAdd(acc + i, alpha * (float)mrow[i]);
  if (l == 0) {
    float c = cwv[e] * alpha;
    atomicAdd(&cacc[d * 3 + 0], (coords[d * 3 + 0] - coords[s * 3 + 0]) * c);
    atomicAdd(&cacc[d * 3 + 1], (coords[d * 3 + 1] - coords[s * 3 + 1]) * c);
    atomicAdd(&cacc[d * 3 + 2], (coords[d * 3 + 2] - coords[s * 3 + 2]) * c);
  }
}

// --------------------------- node MLP + coords update ----------------------
#define LDN 264
__global__ __launch_bounds__(256) void k_node_mlp(
    const _Float16* __restrict__ hbf, const float* __restrict__ hf,
    const float* __restrict__ m_i,
    const _Float16* __restrict__ Wn1, const float* __restrict__ bn1,
    const _Float16* __restrict__ Wn2, const float* __restrict__ bn2,
    int residual, float* __restrict__ hf_n, _Float16* __restrict__ hbf_n,
    const float* __restrict__ coords, const float* __restrict__ cacc,
    float* __restrict__ coords_n, int nN) {
  __shared__ _Float16 nin[64][LDN];
  __shared__ _Float16 t1[64][LDH];
  const int t = threadIdx.x, nb = blockIdx.x * 64;

  { // stage [h | m_i] as f16
    int nl = t >> 2, q = t & 3;
    int node = nb + nl;
    bool ok = node < nN;
    if (ok) {
      const _Float16* hp = hbf + (size_t)node * HID + q * 32;
      cp16(&nin[nl][q * 32], hp);
      cp16(&nin[nl][q * 32 + 8], hp + 8);
      cp16(&nin[nl][q * 32 + 16], hp + 16);
      cp16(&nin[nl][q * 32 + 24], hp + 24);
      const float* mp = m_i + (size_t)node * HID + q * 32;
#pragma unroll
      for (int i = 0; i < 32; ++i)
        nin[nl][HID + q * 32 + i] = (_Float16)mp[i];
    } else {
      uint4 z = {};
#pragma unroll
      for (int c = 0; c < 4; ++c) {
        *(uint4*)&nin[nl][q * 32 + c * 8]       = z;
        *(uint4*)&nin[nl][HID + q * 32 + c * 8] = z;
      }
    }
  }
  cp_fence();
  __syncthreads();

  const int wave = t >> 5, lane = t & 31;
  const int n0 = wave * 16, nn = n0 + (lane & 15), g = lane >> 4;

  for (int rt = 0; rt < 4; ++rt) {   // n1: K=256
    v8f acc = {};
#pragma unroll
    for (int kt = 0; kt < 8; ++kt)
      acc = wmma_f16(frag_a(&nin[rt * 16][kt * 32], LDN, lane),
                     frag_b(Wn1 + (size_t)(kt * 32) * HID + n0, HID, lane), acc);
    float bias = bn1[nn];
#pragma unroll
    for (int r = 0; r < 8; ++r)
      t1[rt * 16 + r + 8 * g][nn] = (_Float16)silu_f(acc[r] + bias);
  }
  __syncthreads();

  for (int rt = 0; rt < 4; ++rt) {   // n2: K=128, + optional residual
    v8f acc = {};
#pragma unroll
    for (int kt = 0; kt < 4; ++kt)
      acc = wmma_f16(frag_a(&t1[rt * 16][kt * 32], LDH, lane),
                     frag_b(Wn2 + (size_t)(kt * 32) * HID + n0, HID, lane), acc);
    float bias = bn2[nn];
#pragma unroll
    for (int r = 0; r < 8; ++r) {
      int node = nb + rt * 16 + r + 8 * g;
      if (node < nN) {
        float v = acc[r] + bias;
        if (residual) v += hf[(size_t)node * HID + nn];
        hf_n[(size_t)node * HID + nn]  = v;
        hbf_n[(size_t)node * HID + nn] = (_Float16)v;
      }
    }
  }
  if (t < 192) {
    int node = nb + t / 3, c = t % 3;
    if (node < nN)
      coords_n[node * 3 + c] = coords[node * 3 + c] + cacc[node * 3 + c];
  }
}

// --------------------------- classifier + output ---------------------------
__global__ __launch_bounds__(256) void k_classifier(
    const _Float16* __restrict__ hbf, const _Float16* __restrict__ Wc1,
    const float* __restrict__ bc1, const float* __restrict__ c2W,
    const float* __restrict__ c2b, const float* __restrict__ coords,
    float* __restrict__ out, int nN) {
  __shared__ _Float16 nin[64][LDH];
  __shared__ _Float16 hid[64][72];
  const int t = threadIdx.x, nb = blockIdx.x * 64;
  {
    int nl = t >> 2, q = t & 3;
    int node = nb + nl;
    if (node < nN) {
      const _Float16* hp = hbf + (size_t)node * HID + q * 32;
      cp16(&nin[nl][q * 32], hp);
      cp16(&nin[nl][q * 32 + 8], hp + 8);
      cp16(&nin[nl][q * 32 + 16], hp + 16);
      cp16(&nin[nl][q * 32 + 24], hp + 24);
    } else {
      uint4 z = {};
#pragma unroll
      for (int c = 0; c < 4; ++c) *(uint4*)&nin[nl][q * 32 + c * 8] = z;
    }
  }
  cp_fence();
  __syncthreads();

  const int wave = t >> 5, lane = t & 31, g = lane >> 4;
  const int ct = wave & 3;  // 4 col tiles (64 cols), 4 row tiles, 2 per wave
  for (int rr = 0; rr < 2; ++rr) {
    int rt = (wave >> 2) * 2 + rr;
    int n0 = ct * 16, nn = n0 + (lane & 15);
    v8f acc = {};
#pragma unroll
    for (int kt = 0; kt < 4; ++kt)
      acc = wmma_f16(frag_a(&nin[rt * 16][kt * 32], LDH, lane),
                     frag_b(Wc1 + (size_t)(kt * 32) * 64 + n0, 64, lane), acc);
    float bias = bc1[nn];
#pragma unroll
    for (int r = 0; r < 8; ++r)
      hid[rt * 16 + r + 8 * g][nn] = (_Float16)silu_f(acc[r] + bias);
  }
  __syncthreads();

  { // final 64-dot per node: 4 lanes/node tree reduce
    int nl = wave * 8 + (lane >> 2);   // 8 nodes per wave -> 64 nodes
    int fp = (lane & 3) * 16;
    float s = 0.0f;
#pragma unroll
    for (int i = 0; i < 16; ++i) s += (float)hid[nl][fp + i] * c2W[fp + i];
#pragma unroll
    for (int m = 2; m > 0; m >>= 1) s += __shfl_xor(s, m, 32);
    if ((lane & 3) == 0) {
      int node = nb + nl;
      if (node < nN) out[node] = s + c2b[0];
    }
  }
  if (t < 192) {
    int node = nb + t / 3, c = t % 3;
    if (node < nN) out[nN + node * 3 + c] = coords[node * 3 + c];
  }
}

// ---------------------------------------------------------------------------
extern "C" void kernel_launch(void* const* d_in, const int* in_sizes, int n_in,
                              void* d_out, int out_size, void* d_ws,
                              size_t ws_size, hipStream_t stream) {
  const int N = N_NODES, E = N_EDGES;
  const float* x     = (const float*)d_in[0];
  const float* cin   = (const float*)d_in[1];
  const int*   eidx  = (const int*)d_in[2];
  const float* eattr = (const float*)d_in[3];

  // Param leaves in jax pytree (sorted-key) order:
  // cls.c1.W, cls.c1.b, cls.c2.W, cls.c2.b, then per layer:
  // att.W, att.b, c1.W, c1.b, c2.W, e1.W, e1.b, e2.W, e2.b, n1.W, n1.b, n2.W, n2.b
  const float* P[56];
  if (n_in >= 60) {
    for (int i = 0; i < 56; ++i) P[i] = (const float*)d_in[4 + i];
  } else {  // single concatenated blob
    static const int lsz[13] = {128, 1, 16384, 128, 128, 33024, 128,
                                16384, 128, 32768, 128, 16384, 128};
    const float* p = (const float*)d_in[4];
    int li = 0;
    int head[4] = {8192, 64, 64, 1};
    for (int i = 0; i < 4; ++i) { P[li++] = p; p += head[i]; }
    for (int l = 0; l < 4; ++l)
      for (int i = 0; i < 13; ++i) { P[li++] = p; p += lsz[i]; }
  }
  auto LP = [&](int l, int k) { return P[4 + l * 13 + k]; };
  const float* clsW1 = P[0]; const float* clsB1 = P[1];
  const float* clsW2 = P[2]; const float* clsB2 = P[3];

  // ---- workspace carve-up ----
  char* ws = (char*)d_ws;
  size_t off = 0;
  auto alloc = [&](size_t bytes) {
    void* p = ws + off;
    off = (off + bytes + 255) & ~(size_t)255;
    return p;
  };
  _Float16* hbf[2]  = {(_Float16*)alloc((size_t)N * HID * 2),
                       (_Float16*)alloc((size_t)N * HID * 2)};
  float* hf[2]      = {(float*)alloc((size_t)N * HID * 4),
                       (float*)alloc((size_t)N * HID * 4)};
  float* cbuf[2]    = {(float*)alloc((size_t)N * 3 * 4),
                       (float*)alloc((size_t)N * 3 * 4)};
  _Float16* m_buf   = (_Float16*)alloc((size_t)E * HID * 2);
  float* expa       = (float*)alloc((size_t)E * 4);
  float* cwv        = (float*)alloc((size_t)E * 4);
  float* dsum       = (float*)alloc((size_t)N * 4);
  float* m_i        = (float*)alloc((size_t)N * HID * 4);
  float* cacc       = (float*)alloc((size_t)N * 3 * 4);
  _Float16 *We1[4], *We2[4], *Wc1l[4], *Wn1[4], *Wn2[4];
  for (int l = 0; l < 4; ++l) {
    We1[l]  = (_Float16*)alloc(288 * HID * 2);
    We2[l]  = (_Float16*)alloc(HID * HID * 2);
    Wc1l[l] = (_Float16*)alloc(HID * HID * 2);
    Wn1[l]  = (_Float16*)alloc(256 * HID * 2);
    Wn2[l]  = (_Float16*)alloc(HID * HID * 2);
  }
  _Float16* Wcls1 = (_Float16*)alloc(HID * 64 * 2);

  auto cvt = [&](const float* s, _Float16* d, int K, int Kpad, int Nc) {
    long long tot = (long long)Kpad * Nc;
    k_cvt_f16<<<(unsigned)((tot + 255) / 256), 256, 0, stream>>>(s, d, K, Kpad,
                                                                 Nc);
  };
  for (int l = 0; l < 4; ++l) {
    cvt(LP(l, 5),  We1[l], 258, 288, HID);
    cvt(LP(l, 7),  We2[l], HID, HID, HID);
    cvt(LP(l, 2),  Wc1l[l], HID, HID, HID);
    cvt(LP(l, 9),  Wn1[l], 256, 256, HID);
    cvt(LP(l, 11), Wn2[l], HID, HID, HID);
  }
  cvt(clsW1, Wcls1, HID, HID, 64);
  cvt(x, hbf[0], N, N, HID);
  (void)hipMemcpyAsync(hf[0], x, (size_t)N * HID * 4,
                       hipMemcpyDeviceToDevice, stream);
  (void)hipMemcpyAsync(cbuf[0], cin, (size_t)N * 3 * 4,
                       hipMemcpyDeviceToDevice, stream);

  int cur = 0;
  for (int l = 0; l < 4; ++l) {
    (void)hipMemsetAsync(dsum, 0, (size_t)N * 4, stream);
    (void)hipMemsetAsync(m_i, 0, (size_t)N * HID * 4, stream);
    (void)hipMemsetAsync(cacc, 0, (size_t)N * 3 * 4, stream);
    k_edge_pass1<<<E / TE, 256, 0, stream>>>(
        hbf[cur], cbuf[cur], eidx, eattr, We1[l], LP(l, 6), We2[l], LP(l, 8),
        Wc1l[l], LP(l, 3), LP(l, 0), LP(l, 1), LP(l, 4), m_buf, expa, cwv,
        dsum, E);
    long long p2 = (long long)E * 32;
    k_edge_pass2<<<(unsigned)((p2 + 255) / 256), 256, 0, stream>>>(
        m_buf, expa, cwv, dsum, eidx, cbuf[cur], m_i, cacc, E);
    int residual = (l > 0 && l < 3) ? 1 : 0;
    k_node_mlp<<<(N + 63) / 64, 256, 0, stream>>>(
        hbf[cur], hf[cur], m_i, Wn1[l], LP(l, 10), Wn2[l], LP(l, 12), residual,
        hf[1 - cur], hbf[1 - cur], cbuf[cur], cacc, cbuf[1 - cur], N);
    cur = 1 - cur;
  }
  k_classifier<<<(N + 63) / 64, 256, 0, stream>>>(
      hbf[cur], Wcls1, clsB1, clsW2, clsB2, cbuf[cur], (float*)d_out, N);
  (void)in_sizes; (void)n_in; (void)out_size; (void)ws_size;
}
